// MultiHeadAttention_1262720385468
// MI455X (gfx1250) — matrix operand order repaired and checked by reading, not compiled, not run on hardware
//
#include <hip/hip_runtime.h>
#include <stdint.h>

#define D_MODEL 1024
#define HEADS   16
#define DK      64
#define NQ      1024
#define NK      1024
#define BATCH   2
#define NTOK    (BATCH * NQ)   // 2048

typedef __attribute__((ext_vector_type(16))) __bf16         v16bf;
typedef __attribute__((ext_vector_type(16))) unsigned short v16u;
typedef __attribute__((ext_vector_type(8)))  unsigned short v8u;
typedef __attribute__((ext_vector_type(8)))  float          v8f;

union Frag {
  v16u  u;
  v16bf b;
  v8u   h[2];
};

__device__ __forceinline__ unsigned short f2bf(float f) {
  unsigned int u = __float_as_uint(f);
  u += 0x7fffu + ((u >> 16) & 1u);      // round-to-nearest-even
  return (unsigned short)(u >> 16);
}

__device__ __forceinline__ v8f wmma_bf16(const Frag& a, const Frag& b, v8f c) {
  // D = A(16x32 bf16) * B(32x16 bf16) + C(16x16 f32)
  return __builtin_amdgcn_wmma_f32_16x16x32_bf16(false, a.b, false, b.b,
                                                 (short)0, c, false, false);
}

// ---------------------------------------------------------------------------
// Kernel 1: W (f32, [K][N] row-major) -> Wt (bf16, [N][K]) via LDS tile
// ---------------------------------------------------------------------------
__global__ __launch_bounds__(256) void transpose_convert_kernel(
    const float* __restrict__ in, unsigned short* __restrict__ out,
    int rows, int cols)
{
  __shared__ float tile[32][33];
  const int c0 = blockIdx.x * 32;
  const int r0 = blockIdx.y * 32;
  const int tx = threadIdx.x;          // 0..31
  const int ty = threadIdx.y;          // 0..7
  #pragma unroll
  for (int i = ty; i < 32; i += 8)
    tile[i][tx] = in[(size_t)(r0 + i) * cols + (c0 + tx)];
  __syncthreads();
  #pragma unroll
  for (int i = ty; i < 32; i += 8)
    out[(size_t)(c0 + i) * rows + (r0 + tx)] = f2bf(tile[tx][i]);
}

// ---------------------------------------------------------------------------
// Kernel 2: fused QKV projection.  grid = (NTOK/64, 48), block = 128 (4 waves)
//   blockIdx.y: [0,16)=Q, [16,32)=K, [32,48)=V ; y&15 = head
//   Q,K stored [b][h][n][64] bf16 ; V stored transposed [b][h][d][n] bf16
//   K-loop is double-buffered so fragment loads overlap the WMMAs.
// ---------------------------------------------------------------------------
__global__ __launch_bounds__(128) void qkv_gemm_kernel(
    const float* __restrict__ Xq, const float* __restrict__ Xk,
    const float* __restrict__ Xv,
    const unsigned short* __restrict__ Wqt, const unsigned short* __restrict__ Wkt,
    const unsigned short* __restrict__ Wvt,
    const float* __restrict__ bq, const float* __restrict__ bk,
    const float* __restrict__ bv,
    unsigned short* __restrict__ Qh, unsigned short* __restrict__ Kh,
    unsigned short* __restrict__ Vt)
{
  const int lane = threadIdx.x & 31;
  const int wave = threadIdx.x >> 5;
  const int lm   = lane & 15;
  const int h2   = lane >> 4;
  const int kbA  = h2 ? 16 : 0;   // A-fragment: 16 consecutive K per half-wave
  const int kbB  = h2 ? 16 : 0;   // B-fragment: 16 consecutive K per half-wave

  const int sel = blockIdx.y >> 4;
  const int hd  = blockIdx.y & 15;
  const float* X = (sel == 0) ? Xq : (sel == 1) ? Xk : Xv;
  const unsigned short* Wt = (sel == 0) ? Wqt : (sel == 1) ? Wkt : Wvt;
  const float* bias = (sel == 0) ? bq : (sel == 1) ? bk : bv;

  const int mBase = blockIdx.x * 64 + wave * 16;
  const int nBase = hd * 64;

  v8f c[4] = {};
  const float* xrow = X + (size_t)(mBase + lm) * D_MODEL;

  auto loadTile = [&](Frag& a, Frag* b, int k0) {
    const float* p = xrow + k0 + kbA;
    float4 x0 = *(const float4*)(p);
    float4 x1 = *(const float4*)(p + 4);
    float4 x2 = *(const float4*)(p + 8);
    float4 x3 = *(const float4*)(p + 12);
    float xs[16] = {x0.x, x0.y, x0.z, x0.w, x1.x, x1.y, x1.z, x1.w,
                    x2.x, x2.y, x2.z, x2.w, x3.x, x3.y, x3.z, x3.w};
    #pragma unroll
    for (int i = 0; i < 16; ++i) a.u[i] = f2bf(xs[i]);
    #pragma unroll
    for (int nt = 0; nt < 4; ++nt)
      b[nt].u = *(const v16u*)(Wt + (size_t)(nBase + nt * 16 + lm) * D_MODEL + k0 + kbB);
  };

  Frag a0, b0[4];
  loadTile(a0, b0, 0);
  for (int k0 = 0; k0 < D_MODEL; k0 += 64) {
    Frag a1, b1[4];
    loadTile(a1, b1, k0 + 32);
    #pragma unroll
    for (int nt = 0; nt < 4; ++nt) c[nt] = wmma_bf16(a0, b0[nt], c[nt]);
    loadTile(a0, b0, (k0 + 64 < D_MODEL) ? (k0 + 64) : 0);  // wrap: dead prefetch
    #pragma unroll
    for (int nt = 0; nt < 4; ++nt) c[nt] = wmma_bf16(a1, b1[nt], c[nt]);
  }

  #pragma unroll
  for (int nt = 0; nt < 4; ++nt) {
    const int d = nt * 16 + lm;
    const float bval = bias[nBase + d];
    if (sel < 2) {
      unsigned short* dst = (sel == 0) ? Qh : Kh;
      #pragma unroll
      for (int r = 0; r < 8; ++r) {
        const int tok = mBase + r + 8 * h2;
        const int bb  = tok >> 10;
        const int n   = tok & 1023;
        dst[((((size_t)bb * HEADS + hd) * NQ + n) << 6) + d] = f2bf(c[nt][r] + bval);
      }
    } else {
      // V transposed: per lane the 8 rows are 8 consecutive tokens -> one 16B store
      v8u pack;
      #pragma unroll
      for (int r = 0; r < 8; ++r) pack[r] = f2bf(c[nt][r] + bval);
      const int tok0 = mBase + 8 * h2;
      const int bb   = tok0 >> 10;
      const int n0   = tok0 & 1023;
      *(v8u*)(Vt + (((size_t)bb * HEADS + hd) * DK + d) * NK + n0) = pack;
    }
  }
}

// ---------------------------------------------------------------------------
// Kernel 3: flash attention.  grid = 256, block = 256 (8 waves).
// Each wave owns one (b, h, 16-query tile); loops over 32-key tiles.
// Software pipelined: K fragments double-buffered across iterations; V
// fragments + weight/mask stream issued at the top of each iteration.
// attO written [token][h*64+d] bf16 (A-layout for output projection).
// ---------------------------------------------------------------------------
#define PSTRIDE 40   // LDS row stride (ushorts): 80B, 16B-aligned, bank-spread

__global__ __launch_bounds__(256) void attn_kernel(
    const unsigned short* __restrict__ Qh, const unsigned short* __restrict__ Kh,
    const unsigned short* __restrict__ Vt,
    const float* __restrict__ attw, const unsigned char* __restrict__ attm,
    unsigned short* __restrict__ attO)
{
  __shared__ __align__(16) unsigned short plds[8][16 * PSTRIDE];

  const int lane = threadIdx.x & 31;
  const int wave = threadIdx.x >> 5;
  const int wid  = blockIdx.x * 8 + wave;       // 0..2047
  const int qt   = wid & 63;
  const int bh   = wid >> 6;                    // b*16+h
  const int hd   = bh & 15;
  const int b    = bh >> 4;
  const int lm   = lane & 15;
  const int h2   = lane >> 4;
  const int kbA  = h2 ? 16 : 0;
  const int kbB  = h2 ? 16 : 0;

  // Q A-fragments (dk 0..31, 32..63), loaded once
  const unsigned short* qrow = Qh + ((size_t)bh * NQ + qt * 16 + lm) * DK;
  Frag aq0, aq1;
  aq0.h[0] = *(const v8u*)(qrow + kbA);
  aq0.h[1] = *(const v8u*)(qrow + kbA + 8);
  aq1.h[0] = *(const v8u*)(qrow + 32 + kbA);
  aq1.h[1] = *(const v8u*)(qrow + 32 + kbA + 8);

  float mi[8], li[8];
  v8f o[4] = {};
  #pragma unroll
  for (int r = 0; r < 8; ++r) { mi[r] = -1e30f; li[r] = 0.f; }

  const size_t wbase = ((size_t)bh * NQ + qt * 16) * (size_t)NK;
  unsigned short* pl = plds[wave];

  // kf[0..1] = dk-chunk 0/1 for key-column-tile 0, kf[2..3] same for tile 1
  auto loadK = [&](Frag* kf, int k0) {
    #pragma unroll
    for (int ct = 0; ct < 2; ++ct) {
      const unsigned short* krow = Kh + ((size_t)bh * NK + k0 + ct * 16 + lm) * DK;
      kf[ct * 2 + 0].u = *(const v16u*)(krow + kbB);
      kf[ct * 2 + 1].u = *(const v16u*)(krow + 32 + kbB);
    }
  };

  auto body = [&](int kt, Frag* kcur, Frag* knxt) {
    const int k0 = kt * 32;

    // ---- issue V fragments + weight/mask stream up front (overlap WMMAs) ----
    Frag vf[4];
    #pragma unroll
    for (int dt = 0; dt < 4; ++dt)
      vf[dt].u = *(const v16u*)(Vt + ((size_t)bh * DK + dt * 16 + lm) * NK + k0 + kbB);

    float w0[8], w1[8];
    unsigned char mk0[8], mk1[8];
    #pragma unroll
    for (int r = 0; r < 8; ++r) {
      const size_t idx = wbase + (size_t)(r + 8 * h2) * NK + k0 + lm;
      w0[r]  = __builtin_nontemporal_load(attw + idx);
      w1[r]  = __builtin_nontemporal_load(attw + idx + 16);
      mk0[r] = __builtin_nontemporal_load(attm + idx);
      mk1[r] = __builtin_nontemporal_load(attm + idx + 16);
    }

    // ---- prefetch next tile's stream + K fragments (double buffer) ----
    const int kn = (k0 + 32 < NK) ? (k0 + 32) : 0;
    __builtin_prefetch(attw + wbase + (size_t)(8 * h2) * NK + kn + lm, 0, 0);
    __builtin_prefetch(attm + wbase + (size_t)(8 * h2) * NK + kn + lm, 0, 0);
    loadK(knxt, kn);

    // ---- S = Q * K^T (two 16-key column tiles) ----
    v8f s[2];
    #pragma unroll
    for (int ct = 0; ct < 2; ++ct) {
      v8f cc = {};
      cc = wmma_bf16(aq0, kcur[ct * 2 + 0], cc);
      cc = wmma_bf16(aq1, kcur[ct * 2 + 1], cc);
      s[ct] = cc;
    }

    // ---- scale * weights, mask, online softmax ----
    float p0[8], p1[8];
    #pragma unroll
    for (int r = 0; r < 8; ++r) {
      float v0 = mk0[r] ? -1e30f : s[0][r] * 0.125f * w0[r];
      float v1 = mk1[r] ? -1e30f : s[1][r] * 0.125f * w1[r];

      float mx = fmaxf(v0, v1);
      mx = fmaxf(mx, __shfl_xor(mx, 1));
      mx = fmaxf(mx, __shfl_xor(mx, 2));
      mx = fmaxf(mx, __shfl_xor(mx, 4));
      mx = fmaxf(mx, __shfl_xor(mx, 8));
      const float nm = fmaxf(mi[r], mx);
      const float sc = __expf(mi[r] - nm);
      const float e0 = __expf(v0 - nm);
      const float e1 = __expf(v1 - nm);
      float rs = e0 + e1;
      rs += __shfl_xor(rs, 1);
      rs += __shfl_xor(rs, 2);
      rs += __shfl_xor(rs, 4);
      rs += __shfl_xor(rs, 8);
      li[r] = li[r] * sc + rs;
      mi[r] = nm;
      #pragma unroll
      for (int dt = 0; dt < 4; ++dt) o[dt][r] *= sc;
      p0[r] = e0;
      p1[r] = e1;
    }

    // ---- P (C-layout) -> LDS -> A-fragment layout (intra-wave transpose) ----
    #pragma unroll
    for (int r = 0; r < 8; ++r) {
      const int qr = r + 8 * h2;
      pl[qr * PSTRIDE + lm]      = f2bf(p0[r]);
      pl[qr * PSTRIDE + 16 + lm] = f2bf(p1[r]);
    }
    __builtin_amdgcn_wave_barrier();
    Frag ap;
    ap.h[0] = *(const v8u*)(pl + lm * PSTRIDE + kbA);
    ap.h[1] = *(const v8u*)(pl + lm * PSTRIDE + kbA + 8);
    __builtin_amdgcn_wave_barrier();

    // ---- O += P * V ----
    #pragma unroll
    for (int dt = 0; dt < 4; ++dt) o[dt] = wmma_bf16(ap, vf[dt], o[dt]);
  };

  Frag kbuf0[4], kbuf1[4];
  loadK(kbuf0, 0);
  for (int kt = 0; kt < NK / 32; kt += 2) {
    body(kt,     kbuf0, kbuf1);
    body(kt + 1, kbuf1, kbuf0);
  }

  float inv[8];
  #pragma unroll
  for (int r = 0; r < 8; ++r) inv[r] = (li[r] > 0.f) ? 1.f / li[r] : 0.f;
  #pragma unroll
  for (int dt = 0; dt < 4; ++dt) {
    #pragma unroll
    for (int r = 0; r < 8; ++r) {
      const int tok = b * NQ + qt * 16 + r + 8 * h2;
      attO[(size_t)tok * D_MODEL + hd * DK + dt * 16 + lm] = f2bf(o[dt][r] * inv[r]);
    }
  }
}

// ---------------------------------------------------------------------------
// Kernel 4: output projection: out_lin = attO(bf16) @ Wo + bo  (f32 out)
// grid = (NTOK/64, D_MODEL/64), block = 128.  Double-buffered K-loop.
// ---------------------------------------------------------------------------
__global__ __launch_bounds__(128) void out_gemm_kernel(
    const unsigned short* __restrict__ A, const unsigned short* __restrict__ Wot,
    const float* __restrict__ bo, float* __restrict__ out)
{
  const int lane = threadIdx.x & 31;
  const int wave = threadIdx.x >> 5;
  const int lm   = lane & 15;
  const int h2   = lane >> 4;
  const int kbA  = h2 ? 16 : 0;
  const int kbB  = h2 ? 16 : 0;

  const int mBase = blockIdx.x * 64 + wave * 16;
  const int nBase = blockIdx.y * 64;

  v8f c[4] = {};
  const unsigned short* arow = A + (size_t)(mBase + lm) * D_MODEL;

  auto loadTile = [&](Frag& a, Frag* b, int k0) {
    a.h[0] = *(const v8u*)(arow + k0 + kbA);
    a.h[1] = *(const v8u*)(arow + k0 + kbA + 8);
    #pragma unroll
    for (int nt = 0; nt < 4; ++nt)
      b[nt].u = *(const v16u*)(Wot + (size_t)(nBase + nt * 16 + lm) * D_MODEL + k0 + kbB);
  };

  Frag a0, b0[4];
  loadTile(a0, b0, 0);
  for (int k0 = 0; k0 < D_MODEL; k0 += 64) {
    Frag a1, b1[4];
    loadTile(a1, b1, k0 + 32);
    #pragma unroll
    for (int nt = 0; nt < 4; ++nt) c[nt] = wmma_bf16(a0, b0[nt], c[nt]);
    loadTile(a0, b0, (k0 + 64 < D_MODEL) ? (k0 + 64) : 0);
    #pragma unroll
    for (int nt = 0; nt < 4; ++nt) c[nt] = wmma_bf16(a1, b1[nt], c[nt]);
  }

  #pragma unroll
  for (int nt = 0; nt < 4; ++nt) {
    const float bval = bo[nBase + nt * 16 + lm];
    #pragma unroll
    for (int r = 0; r < 8; ++r) {
      const int m = mBase + r + 8 * h2;
      out[(size_t)m * D_MODEL + nBase + nt * 16 + lm] = c[nt][r] + bval;
    }
  }
}

// ---------------------------------------------------------------------------
// Kernel 5: y = LayerNorm(queries + out_lin).  One block per token row.
// ---------------------------------------------------------------------------
__global__ __launch_bounds__(256) void ln_kernel(
    const float* __restrict__ q, const float* __restrict__ x,
    const float* __restrict__ gamma, const float* __restrict__ beta,
    float* __restrict__ out)
{
  __shared__ float sdata[16];
  const int row = blockIdx.x;
  const float* qr = q + (size_t)row * D_MODEL;
  const float* xr = x + (size_t)row * D_MODEL;

  float s = 0.f, ss = 0.f;
  for (int i = threadIdx.x; i < D_MODEL; i += 256) {
    const float v = qr[i] + xr[i];
    s += v; ss += v * v;
  }
  #pragma unroll
  for (int m = 1; m < 32; m <<= 1) { s += __shfl_xor(s, m); ss += __shfl_xor(ss, m); }
  const int lane = threadIdx.x & 31, wave = threadIdx.x >> 5;
  if (lane == 0) { sdata[wave] = s; sdata[8 + wave] = ss; }
  __syncthreads();
  if (threadIdx.x < 32) {
    float a  = (threadIdx.x < 8) ? sdata[threadIdx.x] : 0.f;
    float a2 = (threadIdx.x < 8) ? sdata[8 + threadIdx.x] : 0.f;
    #pragma unroll
    for (int m = 1; m < 8; m <<= 1) { a += __shfl_xor(a, m); a2 += __shfl_xor(a2, m); }
    if (threadIdx.x == 0) { sdata[0] = a; sdata[1] = a2; }
  }
  __syncthreads();
  const float mu   = sdata[0] * (1.f / D_MODEL);
  const float var  = sdata[1] * (1.f / D_MODEL) - mu * mu;
  const float rstd = rsqrtf(var + 1e-5f);
  for (int i = threadIdx.x; i < D_MODEL; i += 256) {
    const float v = qr[i] + xr[i];
    out[(size_t)row * D_MODEL + i] = (v - mu) * rstd * gamma[i] + beta[i];
  }
}

// ---------------------------------------------------------------------------
extern "C" void kernel_launch(void* const* d_in, const int* in_sizes, int n_in,
                              void* d_out, int out_size, void* d_ws, size_t ws_size,
                              hipStream_t stream) {
  const float* queries = (const float*)d_in[0];
  const float* keys    = (const float*)d_in[1];
  const float* values  = (const float*)d_in[2];
  const unsigned char* mask = (const unsigned char*)d_in[3];  // numpy bool = 1 byte
  const float* attw    = (const float*)d_in[4];
  const float* Wq = (const float*)d_in[5];  const float* bq = (const float*)d_in[6];
  const float* Wk = (const float*)d_in[7];  const float* bk = (const float*)d_in[8];
  const float* Wv = (const float*)d_in[9];  const float* bv = (const float*)d_in[10];
  const float* Wo = (const float*)d_in[11]; const float* bo = (const float*)d_in[12];
  const float* gamma = (const float*)d_in[13];
  const float* beta  = (const float*)d_in[14];
  float* out = (float*)d_out;

  // workspace layout (~32 MB)
  size_t off = 0;
  char* base = (char*)d_ws;
  auto alloc = [&](size_t bytes) {
    void* p = base + off;
    off += (bytes + 255) & ~(size_t)255;
    return p;
  };
  const size_t wtBytes = (size_t)D_MODEL * D_MODEL * sizeof(unsigned short);
  unsigned short* Wqt  = (unsigned short*)alloc(wtBytes);
  unsigned short* Wkt  = (unsigned short*)alloc(wtBytes);
  unsigned short* Wvt  = (unsigned short*)alloc(wtBytes);
  unsigned short* Wot  = (unsigned short*)alloc(wtBytes);
  unsigned short* Qh   = (unsigned short*)alloc((size_t)BATCH * HEADS * NQ * DK * 2);
  unsigned short* Kh   = (unsigned short*)alloc((size_t)BATCH * HEADS * NK * DK * 2);
  unsigned short* Vt   = (unsigned short*)alloc((size_t)BATCH * HEADS * DK * NK * 2);
  unsigned short* attO = (unsigned short*)alloc((size_t)NTOK * D_MODEL * 2);
  float* out_lin       = (float*)alloc((size_t)NTOK * D_MODEL * 4);
  (void)ws_size; (void)in_sizes; (void)n_in; (void)out_size;

  const dim3 tpBlk(32, 8);
  const dim3 tpGrid(D_MODEL / 32, D_MODEL / 32);
  transpose_convert_kernel<<<tpGrid, tpBlk, 0, stream>>>(Wq, Wqt, D_MODEL, D_MODEL);
  transpose_convert_kernel<<<tpGrid, tpBlk, 0, stream>>>(Wk, Wkt, D_MODEL, D_MODEL);
  transpose_convert_kernel<<<tpGrid, tpBlk, 0, stream>>>(Wv, Wvt, D_MODEL, D_MODEL);
  transpose_convert_kernel<<<tpGrid, tpBlk, 0, stream>>>(Wo, Wot, D_MODEL, D_MODEL);

  qkv_gemm_kernel<<<dim3(NTOK / 64, 48), 128, 0, stream>>>(
      queries, keys, values, Wqt, Wkt, Wvt, bq, bk, bv, Qh, Kh, Vt);

  attn_kernel<<<dim3(BATCH * HEADS * (NQ / 16) / 8), 256, 0, stream>>>(
      Qh, Kh, Vt, attw, mask, attO);

  out_gemm_kernel<<<dim3(NTOK / 64, D_MODEL / 64), 128, 0, stream>>>(
      attO, Wot, bo, out_lin);

  ln_kernel<<<dim3(NTOK), 256, 0, stream>>>(queries, out_lin, gamma, beta, out);
}
